// SummaryNet_5488968204426
// MI455X (gfx1250) — matrix operand order repaired and checked by reading, not compiled, not run on hardware
//
#include <hip/hip_runtime.h>
#include <hip/hip_bf16.h>

// ---------------------------------------------------------------------------
// SummaryNet on MI455X (gfx1250): densify COO weights -> bf16 WMMA GEMMs with
// TDM (tensor_load_to_lds) double-buffered tile staging and fused BN/SiLU.
// ---------------------------------------------------------------------------

typedef __attribute__((ext_vector_type(16))) __bf16   v16bf;
typedef __attribute__((ext_vector_type(8)))  float    v8f;
typedef __attribute__((ext_vector_type(4)))  unsigned v4u;
typedef __attribute__((ext_vector_type(8)))  int      v8i;
typedef __attribute__((ext_vector_type(4)))  int      v4i;

union Frag32B { uint4 u[2]; v16bf v; };

#define BM 128
#define BN 64
#define BK 32
#define LDSS 40   // LDS row stride in bf16 elements (80 bytes, multiple of 16B)

#if defined(__has_builtin)
#if __has_builtin(__builtin_amdgcn_tensor_load_to_lds) && \
    __has_builtin(__builtin_amdgcn_s_wait_tensorcnt)
#define USE_TDM 1
#endif
#endif
#ifndef USE_TDM
#define USE_TDM 0
#endif

__device__ __forceinline__ unsigned short f2bf(float f) {
  unsigned u = __float_as_uint(f);
  u += 0x7FFFu + ((u >> 16) & 1u);           // round-to-nearest-even
  return (unsigned short)(u >> 16);
}

#if USE_TDM
// ---------------------------------------------------------------------------
// Issue one TDM 2D tile load: tile (tile_rows x 32 bf16) from a row-major
// (rows x ld) bf16 tensor at gtile, into LDS rows of stride LDSS (=40 bf16).
// D# per CDNA5 ISA 8.3/8.4: group0 = {flags, lds_addr, global_addr, type=2},
// group1 = {data_size=2B, pad(64B interval, 16B amount), dims, tile, stride}.
// Toolchain here exposes the 6-arg builtin: (g0, g1, g2, g3, g4, cpol).
// ---------------------------------------------------------------------------
__device__ __forceinline__ void tdm_load_tile(const __bf16* gtile,
                                              unsigned lds_byte_addr,
                                              int ld, int rows, int tile_rows) {
  unsigned long long ga = (unsigned long long)(uintptr_t)gtile;
  v4u g0;
  g0.x = 1u;                                     // count=1, user descriptor
  g0.y = lds_byte_addr;                          // [63:32] lds_addr
  g0.z = (unsigned)(ga & 0xffffffffu);           // [95:64] global_addr lo
  g0.w = (unsigned)(ga >> 32) | 0x80000000u;     // [120:96] addr hi, type=2
  v8i g1;
  g1[0] = (1 << 16)      // data_size = 2 bytes
        | (1 << 20)      // pad_enable
        | (3 << 22)      // pad_interval: 16 dwords = 64B (one 32-elem tile row)
        | (3 << 25);     // pad_amount: 4 dwords = 16B  -> LDS row stride 80B
  g1[1] = (ld & 0xffff) << 16;                             // tensor_dim0 lo
  g1[2] = ((ld >> 16) & 0xffff) | ((rows & 0xffff) << 16); // dim0 hi, dim1 lo
  g1[3] = ((rows >> 16) & 0xffff) | (BK << 16);            // dim1 hi, tile_dim0
  g1[4] = tile_rows & 0xffff;                              // tile_dim1
  g1[5] = ld;                                              // tensor_dim0_stride lo
  g1[6] = 0;                                               // stride hi
  g1[7] = 0;
  v4i z4 = {0, 0, 0, 0};
  v8i z8 = {0, 0, 0, 0, 0, 0, 0, 0};
  __builtin_amdgcn_tensor_load_to_lds(g0, g1, z4, z4, z8, 0);
}
#endif

// ---------------------------------------------------------------------------
// COO scatter-add:  W[row*ldw + col] += val   (dense f32 weight build)
// ---------------------------------------------------------------------------
__global__ void scatter_coo(const int* __restrict__ rows,
                            const int* __restrict__ cols,
                            const float* __restrict__ vals,
                            float* __restrict__ W, int ldw, int nnz) {
  int i = blockIdx.x * blockDim.x + threadIdx.x;
  if (i < nnz) {
    atomicAdd(&W[(size_t)rows[i] * (size_t)ldw + (size_t)cols[i]], vals[i]);
  }
}

// ---------------------------------------------------------------------------
// f32 -> packed bf16 conversion (2 elements / thread)
// ---------------------------------------------------------------------------
__global__ void cvt_f32_bf16(const float* __restrict__ src,
                             unsigned* __restrict__ dst, size_t npairs) {
  size_t i = (size_t)blockIdx.x * blockDim.x + threadIdx.x;
  if (i < npairs) {
    float2 f = ((const float2*)src)[i];
    dst[i] = (unsigned)f2bf(f.x) | ((unsigned)f2bf(f.y) << 16);
  }
}

// ---------------------------------------------------------------------------
// bf16 GEMM:  Y[m,n] (+)= sum_k A[m,k] * Wt[n,k]
//   Block tile 128x64, BK=32, 8 waves (4 along M x 2 along N), wave tile 32x32.
//   TDM path: wave 0 DMA-stages both tiles (double buffered, TENSORcnt),
//   all waves compute from LDS with V_WMMA_F32_16X16X32_BF16.
// ---------------------------------------------------------------------------
template <int USE_ATOMIC>
__global__ __launch_bounds__(256) void gemm_bf16_wmma(
    const __bf16* __restrict__ A, int lda,
    const __bf16* __restrict__ Wt, int ldb,
    float* __restrict__ Y, int ldy,
    int ksteps_per_split, int total_ksteps) {
  const int tid  = threadIdx.x;
  const int lane = tid & 31;
  const int wave = tid >> 5;
  const int waveM = wave & 3;
  const int waveN = wave >> 2;
  const int blockN0 = blockIdx.x * BN;
  const int blockM0 = blockIdx.y * BM;

  const int kstep0 = blockIdx.z * ksteps_per_split;
  int kend = kstep0 + ksteps_per_split;
  if (kend > total_ksteps) kend = total_ksteps;

  v8f acc[2][2] = {};
  const int rsel = lane & 15;
  const int hi   = lane >> 4;

#if USE_TDM
  __shared__ __bf16 sA[2][BM * LDSS];
  __shared__ __bf16 sB[2][BN * LDSS];

  const __bf16* gA = A  + (size_t)blockM0 * lda;
  const __bf16* gB = Wt + (size_t)blockN0 * ldb;
  const unsigned ldsA0 = (unsigned)(uintptr_t)&sA[0][0];
  const unsigned ldsA1 = (unsigned)(uintptr_t)&sA[1][0];
  const unsigned ldsB0 = (unsigned)(uintptr_t)&sB[0][0];
  const unsigned ldsB1 = (unsigned)(uintptr_t)&sB[1][0];

  if (wave == 0 && kstep0 < kend) {
    tdm_load_tile(gA + (size_t)kstep0 * BK, ldsA0, lda, BM, BM);
    tdm_load_tile(gB + (size_t)kstep0 * BK, ldsB0, ldb, BN, BN);
  }

  for (int ks = kstep0; ks < kend; ++ks) {
    const int buf = (ks - kstep0) & 1;
    if (wave == 0) {
      if (ks + 1 < kend) {
        tdm_load_tile(gA + (size_t)(ks + 1) * BK, buf ? ldsA0 : ldsA1, lda, BM, BM);
        tdm_load_tile(gB + (size_t)(ks + 1) * BK, buf ? ldsB0 : ldsB1, ldb, BN, BN);
        __builtin_amdgcn_s_wait_tensorcnt(2);   // current stage landed in LDS
      } else {
        __builtin_amdgcn_s_wait_tensorcnt(0);
      }
    }
    __syncthreads();

    Frag32B fa[2], fb[2];
#pragma unroll
    for (int tm = 0; tm < 2; ++tm) {
      const __bf16* p = &sA[buf][(waveM * 32 + tm * 16 + rsel) * LDSS + hi * 8];
      fa[tm].u[0] = *(const uint4*)(p);
      fa[tm].u[1] = *(const uint4*)(p + 16);
    }
#pragma unroll
    for (int tn = 0; tn < 2; ++tn) {
      const __bf16* p = &sB[buf][(waveN * 32 + tn * 16 + rsel) * LDSS + hi * 8];
      fb[tn].u[0] = *(const uint4*)(p);
      fb[tn].u[1] = *(const uint4*)(p + 16);
    }
#pragma unroll
    for (int tm = 0; tm < 2; ++tm)
#pragma unroll
      for (int tn = 0; tn < 2; ++tn)
        acc[tm][tn] = __builtin_amdgcn_wmma_f32_16x16x32_bf16(
            false, fa[tm].v, false, fb[tn].v, (short)0, acc[tm][tn], false, false);

    __syncthreads();   // everyone done reading buf before stage ks+2 overwrites
  }
#else
  __shared__ __bf16 sA[BM * LDSS];
  __shared__ __bf16 sB[BN * LDSS];

  const int rowA = tid >> 1;
  const int colA = (tid & 1) << 4;
  const int rowB = tid >> 2;
  const int colB = (tid & 3) << 3;

  for (int ks = kstep0; ks < kend; ++ks) {
    const int k0 = ks * BK;
    {
      const uint4* src = (const uint4*)(A + (size_t)(blockM0 + rowA) * lda + k0 + colA);
      uint4 a0 = src[0];
      uint4 a1 = src[1];
      uint4* dst = (uint4*)(&sA[rowA * LDSS + colA]);
      dst[0] = a0;
      dst[1] = a1;
    }
    {
      const uint4* src = (const uint4*)(Wt + (size_t)(blockN0 + rowB) * ldb + k0 + colB);
      *(uint4*)(&sB[rowB * LDSS + colB]) = src[0];
    }
    __syncthreads();

    Frag32B fa[2], fb[2];
#pragma unroll
    for (int tm = 0; tm < 2; ++tm) {
      const __bf16* p = &sA[(waveM * 32 + tm * 16 + rsel) * LDSS + hi * 8];
      fa[tm].u[0] = *(const uint4*)(p);
      fa[tm].u[1] = *(const uint4*)(p + 16);
    }
#pragma unroll
    for (int tn = 0; tn < 2; ++tn) {
      const __bf16* p = &sB[(waveN * 32 + tn * 16 + rsel) * LDSS + hi * 8];
      fb[tn].u[0] = *(const uint4*)(p);
      fb[tn].u[1] = *(const uint4*)(p + 16);
    }
#pragma unroll
    for (int tm = 0; tm < 2; ++tm)
#pragma unroll
      for (int tn = 0; tn < 2; ++tn)
        acc[tm][tn] = __builtin_amdgcn_wmma_f32_16x16x32_bf16(
            false, fa[tm].v, false, fb[tn].v, (short)0, acc[tm][tn], false, false);

    __syncthreads();
  }
#endif

  // C/D layout: lane L -> n = L&15 ; VGPR v -> m = v + (L>>4)*8
  const int mb = blockM0 + waveM * 32;
  const int nb = blockN0 + waveN * 32;
#pragma unroll
  for (int tm = 0; tm < 2; ++tm) {
#pragma unroll
    for (int tn = 0; tn < 2; ++tn) {
      float* p = Y + (size_t)(mb + tm * 16 + hi * 8) * ldy + (nb + tn * 16 + rsel);
#pragma unroll
      for (int v = 0; v < 8; ++v, p += ldy) {
        if (USE_ATOMIC) atomicAdd(p, acc[tm][tn][v]);
        else            *p = acc[tm][tn][v];
      }
    }
  }
}

// ---------------------------------------------------------------------------
// BatchNorm batch statistics: per-column mean and rsqrt(var+eps) over M rows
// ---------------------------------------------------------------------------
__global__ void bn_stats(const float* __restrict__ Y, int ldy, int M,
                         float* __restrict__ mean, float* __restrict__ rvar, int N) {
  int n = blockIdx.x * blockDim.x + threadIdx.x;
  if (n >= N) return;
  float s = 0.f, s2 = 0.f;
  for (int r = 0; r < M; ++r) {
    float v = Y[(size_t)r * ldy + n];
    s += v; s2 += v * v;
  }
  float m = s / (float)M;
  float var = s2 / (float)M - m * m;
  mean[n] = m;
  rvar[n] = rsqrtf(var + 1e-5f);
}

// ---------------------------------------------------------------------------
// h = bf16( silu( (y - mean) * rvar * g + be ) )   (bias absorbed by BN)
// ---------------------------------------------------------------------------
__global__ void bn_silu_bf16(const float* __restrict__ Y, int ldy,
                             const float* __restrict__ mean,
                             const float* __restrict__ rvar,
                             const float* __restrict__ g,
                             const float* __restrict__ be,
                             unsigned short* __restrict__ H, int ldh,
                             int M, int N) {
  int idx = blockIdx.x * blockDim.x + threadIdx.x;
  if (idx >= M * N) return;
  int r = idx / N, n = idx - r * N;
  float t = (Y[(size_t)r * ldy + n] - mean[n]) * rvar[n] * g[n] + be[n];
  float h = t / (1.f + __expf(-t));
  H[(size_t)r * ldh + n] = f2bf(h);
}

// ---------------------------------------------------------------------------
// h = bf16( silu( y + bias ) )    (layer 2: no BN)
// ---------------------------------------------------------------------------
__global__ void bias_silu_bf16(const float* __restrict__ Y, int ldy,
                               const float* __restrict__ bias,
                               unsigned short* __restrict__ H, int ldh,
                               int M, int N) {
  int idx = blockIdx.x * blockDim.x + threadIdx.x;
  if (idx >= M * N) return;
  int r = idx / N, n = idx - r * N;
  float t = Y[(size_t)r * ldy + n] + bias[n];
  float h = t / (1.f + __expf(-t));
  H[(size_t)r * ldh + n] = f2bf(h);
}

// ---------------------------------------------------------------------------
// out[b, n] = y5[b, n] + b5[n]   (final layer, f32, unpadded 512 x 300)
// ---------------------------------------------------------------------------
__global__ void final_bias(const float* __restrict__ Y, int ldy,
                           const float* __restrict__ bias,
                           float* __restrict__ out, int M, int N) {
  int idx = blockIdx.x * blockDim.x + threadIdx.x;
  if (idx >= M * N) return;
  int r = idx / N, n = idx - r * N;
  out[idx] = Y[(size_t)r * ldy + n] + bias[n];
}

// ---------------------------------------------------------------------------
// Host-side orchestration
// ---------------------------------------------------------------------------
static inline size_t alignUp(size_t x) { return (x + 255) & ~(size_t)255; }

extern "C" void kernel_launch(void* const* d_in, const int* in_sizes, int n_in,
                              void* d_out, int out_size, void* d_ws, size_t ws_size,
                              hipStream_t stream) {
  (void)n_in; (void)out_size; (void)ws_size;

  const int Bsz = 512;
  const int Kd  = 120000;
  const int P1  = 640;   // 600 padded to multiple of 64
  const int P2  = 320;   // 300 padded to multiple of 64

  // ---- inputs ----
  const float* x  = (const float*)d_in[0];
  const int*   idx[5]; const float* val[5]; const float* bia[5];
  int nnz[5];
  for (int i = 0; i < 5; ++i) {
    idx[i] = (const int*)d_in[1 + 3 * i];
    val[i] = (const float*)d_in[2 + 3 * i];
    bia[i] = (const float*)d_in[3 + 3 * i];
    nnz[i] = in_sizes[1 + 3 * i] / 2;
  }
  const float* g1  = (const float*)d_in[16];
  const float* be1 = (const float*)d_in[17];
  const float* g2  = (const float*)d_in[18];
  const float* be2 = (const float*)d_in[19];
  const float* g3  = (const float*)d_in[20];
  const float* be3 = (const float*)d_in[21];

  const int Ntru[5] = {600, 600, 300, 300, 300};
  const int Npad[5] = {P1,  P1,  P2,  P2,  P2 };
  const int Kpad[5] = {Kd,  P1,  P1,  P2,  P2 };

  // ---- workspace layout ----
  char* ws = (char*)d_ws;
  size_t off = 0;
  auto carve = [&](size_t bytes) { size_t o = off; off = alignUp(off + bytes); return o; };

  size_t oWf[5], oWb[5];
  for (int i = 0; i < 5; ++i) oWf[i] = carve((size_t)Npad[i] * Kpad[i] * 4);
  for (int i = 0; i < 5; ++i) oWb[i] = carve((size_t)Npad[i] * Kpad[i] * 2);
  size_t oXb = carve((size_t)Bsz * Kd * 2);
  size_t oY[5];
  for (int i = 0; i < 5; ++i) oY[i] = carve((size_t)Bsz * Npad[i] * 4);
  size_t oH[4];
  for (int i = 0; i < 4; ++i) oH[i] = carve((size_t)Bsz * Npad[i] * 2);
  size_t oMean = carve((size_t)P1 * 4);
  size_t oRvar = carve((size_t)P1 * 4);

  float*  Wf[5]; __bf16* Wb[5];
  for (int i = 0; i < 5; ++i) { Wf[i] = (float*)(ws + oWf[i]); Wb[i] = (__bf16*)(ws + oWb[i]); }
  __bf16* Xb = (__bf16*)(ws + oXb);
  float*  Y[5];  for (int i = 0; i < 5; ++i) Y[i] = (float*)(ws + oY[i]);
  unsigned short* H[4]; for (int i = 0; i < 4; ++i) H[i] = (unsigned short*)(ws + oH[i]);
  float* dMean = (float*)(ws + oMean);
  float* dRvar = (float*)(ws + oRvar);

  // ---- 1. zero dense weights (padding stays 0), Y1 (split-K atomics), H ----
  for (int i = 0; i < 5; ++i)
    (void)hipMemsetAsync(Wf[i], 0, (size_t)Npad[i] * Kpad[i] * 4, stream);
  (void)hipMemsetAsync(Y[0], 0, (size_t)Bsz * Npad[0] * 4, stream);
  for (int i = 0; i < 4; ++i)
    (void)hipMemsetAsync(H[i], 0, (size_t)Bsz * Npad[i] * 2, stream);

  // ---- 2. densify COO weights ----
  for (int i = 0; i < 5; ++i) {
    int blocks = (nnz[i] + 255) / 256;
    scatter_coo<<<blocks, 256, 0, stream>>>(idx[i], idx[i] + nnz[i], val[i],
                                            Wf[i], Kpad[i], nnz[i]);
  }

  // ---- 3. convert weights and x to bf16 ----
  for (int i = 0; i < 5; ++i) {
    size_t pairs = (size_t)Npad[i] * Kpad[i] / 2;
    cvt_f32_bf16<<<(unsigned)((pairs + 255) / 256), 256, 0, stream>>>(
        Wf[i], (unsigned*)Wb[i], pairs);
  }
  {
    size_t pairs = (size_t)Bsz * Kd / 2;
    cvt_f32_bf16<<<(unsigned)((pairs + 255) / 256), 256, 0, stream>>>(
        x, (unsigned*)Xb, pairs);
  }

  // ---- 4. layer 1: WMMA GEMM with split-K + BN + SiLU ----
  {
    const int totalK = Kd / BK;              // 3750 k-steps
    const int splitK = 16;
    const int kper   = (totalK + splitK - 1) / splitK;
    dim3 grid(P1 / BN, Bsz / BM, splitK);
    gemm_bf16_wmma<1><<<grid, 256, 0, stream>>>(Xb, Kd, Wb[0], Kd, Y[0], P1,
                                                kper, totalK);
    bn_stats<<<(Ntru[0] + 255) / 256, 256, 0, stream>>>(Y[0], P1, Bsz, dMean, dRvar, Ntru[0]);
    bn_silu_bf16<<<(Bsz * Ntru[0] + 255) / 256, 256, 0, stream>>>(
        Y[0], P1, dMean, dRvar, g1, be1, H[0], P1, Bsz, Ntru[0]);
  }

  // ---- 5. layer 2: GEMM + bias + SiLU (no BN) ----
  {
    const int totalK = P1 / BK;
    dim3 grid(P1 / BN, Bsz / BM, 1);
    gemm_bf16_wmma<0><<<grid, 256, 0, stream>>>((const __bf16*)H[0], P1, Wb[1], P1,
                                                Y[1], P1, totalK, totalK);
    bias_silu_bf16<<<(Bsz * Ntru[1] + 255) / 256, 256, 0, stream>>>(
        Y[1], P1, bia[1], H[1], P1, Bsz, Ntru[1]);
  }

  // ---- 6. layer 3: GEMM + BN + SiLU ----
  {
    const int totalK = P1 / BK;
    dim3 grid(P2 / BN, Bsz / BM, 1);
    gemm_bf16_wmma<0><<<grid, 256, 0, stream>>>((const __bf16*)H[1], P1, Wb[2], P1,
                                                Y[2], P2, totalK, totalK);
    bn_stats<<<(Ntru[2] + 255) / 256, 256, 0, stream>>>(Y[2], P2, Bsz, dMean, dRvar, Ntru[2]);
    bn_silu_bf16<<<(Bsz * Ntru[2] + 255) / 256, 256, 0, stream>>>(
        Y[2], P2, dMean, dRvar, g2, be2, H[2], P2, Bsz, Ntru[2]);
  }

  // ---- 7. layer 4: GEMM + BN + SiLU ----
  {
    const int totalK = P2 / BK;
    dim3 grid(P2 / BN, Bsz / BM, 1);
    gemm_bf16_wmma<0><<<grid, 256, 0, stream>>>((const __bf16*)H[2], P2, Wb[3], P2,
                                                Y[3], P2, totalK, totalK);
    bn_stats<<<(Ntru[3] + 255) / 256, 256, 0, stream>>>(Y[3], P2, Bsz, dMean, dRvar, Ntru[3]);
    bn_silu_bf16<<<(Bsz * Ntru[3] + 255) / 256, 256, 0, stream>>>(
        Y[3], P2, dMean, dRvar, g3, be3, H[3], P2, Bsz, Ntru[3]);
  }

  // ---- 8. layer 5: GEMM + bias -> d_out (512 x 300 f32) ----
  {
    const int totalK = P2 / BK;
    dim3 grid(P2 / BN, Bsz / BM, 1);
    gemm_bf16_wmma<0><<<grid, 256, 0, stream>>>((const __bf16*)H[3], P2, Wb[4], P2,
                                                Y[4], P2, totalK, totalK);
    final_bias<<<(Bsz * Ntru[4] + 255) / 256, 256, 0, stream>>>(
        Y[4], P2, bia[4], (float*)d_out, Bsz, Ntru[4]);
  }
}
